// LinearAttention_88450556493920
// MI455X (gfx1250) — matrix operand order repaired and checked by reading, not compiled, not run on hardware
//
#include <hip/hip_runtime.h>
#include <math.h>

typedef __attribute__((ext_vector_type(16))) _Float16 v16h;
typedef __attribute__((ext_vector_type(2)))  __fp16   v2fp16;
typedef __attribute__((ext_vector_type(8)))  float    v8f;

#define N_B    4
#define L_SEQ  4096
#define S_SEQ  4096
#define E_DIM  1024
#define H_NUM  16
#define D_HEAD 64
#define EPS_LA 1e-6f

__device__ __forceinline__ float elu1(float x) {
    // jax.nn.elu(x) + 1  ==  x>0 ? x+1 : exp(x)
    return x > 0.0f ? x + 1.0f : __expf(x);
}

__device__ __forceinline__ unsigned pk2h(float a, float b) {
    union { v2fp16 h; unsigned u; } cv;
    cv.h = __builtin_amdgcn_cvt_pkrtz(a, b);  // v_cvt_pk_rtz_f16_f32
    return cv.u;
}

__device__ __forceinline__ uint2 pk4h(const float4& f) {
    return make_uint2(pk2h(f.x, f.y), pk2h(f.z, f.w));
}

// C[M, Nout] = act( A[M,K] @ W[Nout,K]^T + bias )
// fp32 in/out, f16 WMMA with fp32 accumulation.
// Block tile 64(M) x 128(N), K-chunk 64, double-buffered LDS, 256 thr = 8 waves,
// each wave owns a 32x32 tile = 2x2 v_wmma_f32_16x16x32_f16 per 32-K step.
template <int ACT>
__global__ __launch_bounds__(256) void gemm_wmma_f16(
    const float* __restrict__ A, const float* __restrict__ W,
    const float* __restrict__ bias, float* __restrict__ C,
    int M, int K, int Nout)
{
    constexpr int BM = 64, BN = 128, BK = 64;
    constexpr int LDA = BK + 8;  // 72 halves = 144B row stride (16B multiple)
    __shared__ __align__(16) _Float16 As[2][BM][LDA];
    __shared__ __align__(16) _Float16 Bs[2][BN][LDA];

    const int tid  = threadIdx.x;
    const int lane = tid & 31;
    const int wave = tid >> 5;
    const int wm   = wave >> 2;   // 0..1 : 32 rows
    const int wn   = wave & 3;    // 0..3 : 32 cols
    const int lm   = lane & 15;
    const int lh   = lane >> 4;

    const int rowBase = blockIdx.y * BM;
    const int colBase = blockIdx.x * BN;
    const int lr = tid >> 4;      // 0..15 load row group
    const int lc = tid & 15;      // vec4 column 0..15

    v8f acc[2][2] = {};
    float4 areg[4], breg[8];

    auto loadRegs = [&](int k0) {
#pragma unroll
        for (int i = 0; i < 4; ++i)
            areg[i] = *(const float4*)(A + (size_t)(rowBase + lr + i * 16) * K + k0 + lc * 4);
#pragma unroll
        for (int i = 0; i < 8; ++i)
            breg[i] = *(const float4*)(W + (size_t)(colBase + lr + i * 16) * K + k0 + lc * 4);
    };
    auto storeLds = [&](int buf) {
#pragma unroll
        for (int i = 0; i < 4; ++i)
            *(uint2*)&As[buf][lr + i * 16][lc * 4] = pk4h(areg[i]);
#pragma unroll
        for (int i = 0; i < 8; ++i)
            *(uint2*)&Bs[buf][lr + i * 16][lc * 4] = pk4h(breg[i]);
    };
    auto compute = [&](int buf) {
#pragma unroll
        for (int ks = 0; ks < BK; ks += 32) {
#pragma unroll
            for (int mt = 0; mt < 2; ++mt) {
                // A frag 16x32: lane row M=lm; elem i -> K = ks + 16*(i>>3) + 8*lh + (i&7)
                v16h afrag;
                const _Float16* arow = &As[buf][wm * 32 + mt * 16 + lm][ks];
#pragma unroll
                for (int i = 0; i < 16; ++i)
                    afrag[i] = arow[16 * (i >> 3) + 8 * lh + (i & 7)];
#pragma unroll
                for (int nt = 0; nt < 2; ++nt) {
                    // B frag 32x16: lane col N=lm; elem i -> K = ks + 16*(i>>3) + 8*lh + (i&7)
                    v16h bfrag;
                    const _Float16* brow = &Bs[buf][wn * 32 + nt * 16 + lm][ks];
#pragma unroll
                    for (int i = 0; i < 16; ++i) bfrag[i] = brow[16 * (i >> 3) + 8 * lh + (i & 7)];
                    acc[mt][nt] = __builtin_amdgcn_wmma_f32_16x16x32_f16(
                        false, afrag, false, bfrag, (short)0, acc[mt][nt], false, false);
                }
            }
        }
    };

    // Software pipeline: global loads for chunk k+1 issue before compute of chunk k.
    loadRegs(0);
    storeLds(0);
    __syncthreads();
    const int nk = K / BK;
    for (int kb = 0; kb < nk; ++kb) {
        if (kb + 1 < nk) loadRegs((kb + 1) * BK);
        compute(kb & 1);
        if (kb + 1 < nk) storeLds((kb + 1) & 1);
        __syncthreads();
    }

    // C/D layout: elem j -> (M = 8*lh + j, N = lm) within 16x16 tile
#pragma unroll
    for (int mt = 0; mt < 2; ++mt) {
#pragma unroll
        for (int nt = 0; nt < 2; ++nt) {
            int n  = colBase + wn * 32 + nt * 16 + lm;
            int m0 = rowBase + wm * 32 + mt * 16 + 8 * lh;
            float b = bias[n];
#pragma unroll
            for (int j = 0; j < 8; ++j) {
                float v = acc[mt][nt][j] + b;
                if (ACT) v = elu1(v);
                C[(size_t)(m0 + j) * Nout + n] = v;
            }
        }
    }
}

// KV[n,h,m,d] = sum_s V[n,s,h,m] * K[n,s,h,d];  Ksum[n,h,d] = sum_s K[n,s,h,d]
// One block per (n,h). fp32 tiles staged via CDNA5 async global->LDS copies.
__global__ __launch_bounds__(256) void kv_state_kernel(
    const float* __restrict__ Kf, const float* __restrict__ Vf,
    float* __restrict__ KV, float* __restrict__ Ksum)
{
    constexpr int TS = 32;
    __shared__ __align__(16) float Kt[TS][D_HEAD];
    __shared__ __align__(16) float Vt[TS][D_HEAD];

    const int nh  = blockIdx.x;
    const int n   = nh / H_NUM;
    const int h   = nh % H_NUM;
    const int tid = threadIdx.x;
    const int d   = tid & 63;
    const int mg  = tid >> 6;  // 0..3

    float acc[16] = {};
    float ks = 0.0f;

    const size_t base = (size_t)n * S_SEQ * E_DIM + (size_t)h * D_HEAD;
    for (int s0 = 0; s0 < S_SEQ; s0 += TS) {
        // Async copy 32x64 fp32 K and V tiles into LDS (b128 per lane).
#pragma unroll
        for (int i = 0; i < 2; ++i) {
            int idx = tid + i * 256;
            int ss = idx >> 4, c4 = idx & 15;
            const float* gK = Kf + base + (size_t)(s0 + ss) * E_DIM + c4 * 4;
            const float* gV = Vf + base + (size_t)(s0 + ss) * E_DIM + c4 * 4;
            unsigned lK = (unsigned)(size_t)&Kt[ss][c4 * 4];
            unsigned lV = (unsigned)(size_t)&Vt[ss][c4 * 4];
            asm volatile("global_load_async_to_lds_b128 %0, %1, off"
                         :: "v"(lK), "v"(gK) : "memory");
            asm volatile("global_load_async_to_lds_b128 %0, %1, off"
                         :: "v"(lV), "v"(gV) : "memory");
        }
        asm volatile("s_wait_asynccnt 0x0" ::: "memory");
        __syncthreads();

#pragma unroll 4
        for (int ss = 0; ss < TS; ++ss) {
            float kd = Kt[ss][d];
#pragma unroll
            for (int i = 0; i < 16; ++i)
                acc[i] = fmaf(Vt[ss][mg * 16 + i], kd, acc[i]);
            if (mg == 0) ks += kd;
        }
        __syncthreads();
    }

    float* kvb = KV + (size_t)nh * D_HEAD * D_HEAD;
#pragma unroll
    for (int i = 0; i < 16; ++i)
        kvb[(size_t)(mg * 16 + i) * D_HEAD + d] = acc[i];
    if (mg == 0) Ksum[(size_t)nh * D_HEAD + d] = ks;
}

// Vo[n,l,h*64+m] = (sum_d Q[n,l,h,d] * KV[n,h,m,d]) * Z(n,l,h)
__global__ __launch_bounds__(256) void apply_kernel(
    const float* __restrict__ Q, const float* __restrict__ KV,
    const float* __restrict__ Ksum, float* __restrict__ Vo)
{
    __shared__ __align__(16) float Qs[E_DIM];
    __shared__ float Zs[H_NUM];

    const int nl  = blockIdx.x;
    const int n   = nl / L_SEQ;
    const int tid = threadIdx.x;

    const float4* qrow4 = (const float4*)(Q + (size_t)nl * E_DIM);
    ((float4*)Qs)[tid] = qrow4[tid];  // 256 * float4 = 1024 floats
    __syncthreads();

    if (tid < H_NUM) {
        const float4* ks4 = (const float4*)(Ksum + ((size_t)n * H_NUM + tid) * D_HEAD);
        const float4* q4  = (const float4*)&Qs[tid * D_HEAD];
        float z = 0.0f;
#pragma unroll
        for (int dd = 0; dd < D_HEAD / 4; ++dd) {
            float4 a = q4[dd], b = ks4[dd];
            z = fmaf(a.x, b.x, fmaf(a.y, b.y, fmaf(a.z, b.z, fmaf(a.w, b.w, z))));
        }
        Zs[tid] = 1.0f / (z + EPS_LA);
    }
    __syncthreads();

    float* orow = Vo + (size_t)nl * E_DIM;
#pragma unroll
    for (int i = 0; i < E_DIM / 256; ++i) {
        int e = tid + i * 256;
        int hh = e >> 6, m = e & 63;
        const float4* kv4 = (const float4*)(KV + (((size_t)n * H_NUM + hh) * D_HEAD + m) * D_HEAD);
        const float4* q4  = (const float4*)&Qs[hh * D_HEAD];
        float a = 0.0f;
#pragma unroll
        for (int dd = 0; dd < D_HEAD / 4; ++dd) {
            float4 x = q4[dd], y = kv4[dd];
            a = fmaf(x.x, y.x, fmaf(x.y, y.y, fmaf(x.z, y.z, fmaf(x.w, y.w, a))));
        }
        orow[e] = a * Zs[hh];
    }
}

extern "C" void kernel_launch(void* const* d_in, const int* in_sizes, int n_in,
                              void* d_out, int out_size, void* d_ws, size_t ws_size,
                              hipStream_t stream) {
    const float* query = (const float*)d_in[0];
    const float* key   = (const float*)d_in[1];
    const float* value = (const float*)d_in[2];
    const float* Wq    = (const float*)d_in[3];
    const float* bq    = (const float*)d_in[4];
    const float* Wk    = (const float*)d_in[5];
    const float* bk    = (const float*)d_in[6];
    const float* Wv    = (const float*)d_in[7];
    const float* bv    = (const float*)d_in[8];
    const float* Wp    = (const float*)d_in[9];
    const float* bp    = (const float*)d_in[10];

    float* ws = (float*)d_ws;
    const size_t tok = (size_t)N_B * L_SEQ * E_DIM;
    float* Qb  = ws;
    float* Kb  = Qb + tok;
    float* Vb  = Kb + tok;
    float* Ab  = Vb + tok;
    float* KVb = Ab + tok;
    float* KSb = KVb + (size_t)N_B * H_NUM * D_HEAD * D_HEAD;

    const int M = N_B * L_SEQ;  // 16384
    dim3 gGemm(E_DIM / 128, M / 64, 1);
    dim3 b256(256, 1, 1);

    gemm_wmma_f16<1><<<gGemm, b256, 0, stream>>>(query, Wq, bq, Qb, M, E_DIM, E_DIM);
    gemm_wmma_f16<1><<<gGemm, b256, 0, stream>>>(key,   Wk, bk, Kb, M, E_DIM, E_DIM);
    gemm_wmma_f16<0><<<gGemm, b256, 0, stream>>>(value, Wv, bv, Vb, M, E_DIM, E_DIM);

    kv_state_kernel<<<dim3(N_B * H_NUM), b256, 0, stream>>>(Kb, Vb, KVb, KSb);
    apply_kernel<<<dim3(N_B * L_SEQ), b256, 0, stream>>>(Qb, KVb, KSb, Ab);

    gemm_wmma_f16<0><<<gGemm, b256, 0, stream>>>(Ab, Wp, bp, (float*)d_out, M, E_DIM, E_DIM);
}
